// LlamaAttention_43173011259424
// MI455X (gfx1250) — matrix operand order repaired and checked by reading, not compiled, not run on hardware
//
#include <hip/hip_runtime.h>
#include <stdint.h>

// ---- problem constants -----------------------------------------------------
#define BATCH 2
#define SEQ 2048
#define DIM 2048
#define NUM_HEADS 32
#define NUM_KV_HEADS 8
#define HEAD_DIM 64
#define KV_DIM 512
#define QKV_N (DIM + 2 * KV_DIM)   // 3072: [Q | K | V] per token row
#define TOKENS (BATCH * SEQ)       // 4096

typedef __attribute__((ext_vector_type(16))) __bf16 v16bf;
typedef __attribute__((ext_vector_type(8)))  float  v8f;

union TileU { uint4 q[2]; v16bf v; };

__device__ __forceinline__ uint16_t f2bf(float f) {
  union { float f; uint32_t u; } x; x.f = f;
  uint32_t u = x.u;
  u += 0x7FFFu + ((u >> 16) & 1u);   // round-to-nearest-even
  return (uint16_t)(u >> 16);
}

__device__ __forceinline__ v8f wmma_bf16(v16bf a, v16bf b, v8f c) {
  return __builtin_amdgcn_wmma_f32_16x16x32_bf16(false, a, false, b, (short)0, c,
                                                 false, false);
}

// A-tile (16x32, row-major source, ld in elements):
// lane L: row = L&15, elems 0..7 = K[bk..bk+7], 8..15 = K[bk+16..bk+23], bk=(L>>4)*8
__device__ __forceinline__ v16bf load_a_tile(const uint16_t* __restrict__ base, int ld) {
  const int lane = threadIdx.x & 31;
  const int r  = lane & 15;
  const int ks = (lane >> 4) << 3;
  const uint16_t* p = base + (size_t)r * ld + ks;
  TileU t;
  t.q[0] = *(const uint4*)p;
  t.q[1] = *(const uint4*)(p + 16);
  return t.v;
}

// B-tile (32x16) taken from Bt = B^T row-major [N x K] (ld in elements):
// lane L: col n = L&15, elems 0..7 = K[bk..bk+7], 8..15 = K[bk+16..bk+23], bk=(L>>4)*8
__device__ __forceinline__ v16bf load_b_tile(const uint16_t* __restrict__ bt, int ld) {
  const int lane = threadIdx.x & 31;
  const int n  = lane & 15;
  const int ks = (lane >> 4) << 3;
  const uint16_t* p = bt + (size_t)n * ld + ks;
  TileU t;
  t.q[0] = *(const uint4*)p;
  t.q[1] = *(const uint4*)(p + 16);
  return t.v;
}

// ---- prep kernels ----------------------------------------------------------
__global__ void convert_f32_bf16(const float* __restrict__ in,
                                 uint16_t* __restrict__ out, int n) {
  int i = (blockIdx.x * blockDim.x + threadIdx.x) * 4;
  if (i >= n) return;
  float4 v = *(const float4*)(in + i);
  uint16_t* o = out + i;
  o[0] = f2bf(v.x); o[1] = f2bf(v.y); o[2] = f2bf(v.z); o[3] = f2bf(v.w);
}

// in: f32 [R x C] row-major -> out: bf16 [C x R] row-major (i.e. transposed)
__global__ __launch_bounds__(256) void transpose_w(const float* __restrict__ in,
                                                   uint16_t* __restrict__ out,
                                                   int R, int C) {
  __shared__ float tile[32][33];
  const int c0 = blockIdx.x * 32, r0 = blockIdx.y * 32;
  const int tx = threadIdx.x, ty = threadIdx.y;  // (32, 8)
#pragma unroll
  for (int i = 0; i < 32; i += 8)
    tile[ty + i][tx] = in[(size_t)(r0 + ty + i) * C + (c0 + tx)];
  __syncthreads();
#pragma unroll
  for (int i = 0; i < 32; i += 8)
    out[(size_t)(c0 + ty + i) * R + (r0 + tx)] = f2bf(tile[tx][ty + i]);
}

// Vt[(b*KVH + kvh)*64 + d][s] = QKV[b*SEQ + s][2560 + kvh*64 + d]
__global__ void build_vt(const uint16_t* __restrict__ qkv,
                         uint16_t* __restrict__ vt) {
  int idx = blockIdx.x * blockDim.x + threadIdx.x;  // 2*8*64*2048 total
  int s   = idx & (SEQ - 1);
  int d   = (idx >> 11) & 63;
  int kvh = (idx >> 17) & 7;
  int b   = idx >> 20;
  vt[idx] = qkv[(size_t)(b * SEQ + s) * QKV_N + (DIM + KV_DIM) + kvh * 64 + d];
}

// ---- bf16 GEMM: C[MxN] = A[MxK] * Bt[NxK]^T, f32 accumulate ----------------
// 64x64 tile per wave; register double-buffer so the next K-step's 16 b128
// loads are in flight behind the current step's 16 WMMAs.
template <bool F32OUT>
__global__ __launch_bounds__(32) void gemm_bf16(const uint16_t* __restrict__ A,
                                                const uint16_t* __restrict__ Bt,
                                                void* __restrict__ Cout,
                                                int M, int N, int K) {
  const int j0 = blockIdx.x * 64;
  const int i0 = blockIdx.y * 64;
  v8f c[4][4];
#pragma unroll
  for (int i = 0; i < 4; ++i)
#pragma unroll
    for (int j = 0; j < 4; ++j) c[i][j] = (v8f){};

  const uint16_t* ap[4];
  const uint16_t* bp[4];
#pragma unroll
  for (int i = 0; i < 4; ++i) {
    ap[i] = A + (size_t)(i0 + i * 16) * K;
    bp[i] = Bt + (size_t)(j0 + i * 16) * K;
  }

  v16bf a[4], b[4], an[4], bn[4];
#pragma unroll
  for (int i = 0; i < 4; ++i) a[i] = load_a_tile(ap[i], K);
#pragma unroll
  for (int j = 0; j < 4; ++j) b[j] = load_b_tile(bp[j], K);

  for (int k = 32; k < K; k += 32) {
    // prefetch next K-step into the shadow buffer
#pragma unroll
    for (int i = 0; i < 4; ++i) an[i] = load_a_tile(ap[i] + k, K);
#pragma unroll
    for (int j = 0; j < 4; ++j) bn[j] = load_b_tile(bp[j] + k, K);
    // 16 WMMAs on the current buffer while the loads are in flight
#pragma unroll
    for (int i = 0; i < 4; ++i)
#pragma unroll
      for (int j = 0; j < 4; ++j) c[i][j] = wmma_bf16(a[i], b[j], c[i][j]);
#pragma unroll
    for (int i = 0; i < 4; ++i) { a[i] = an[i]; b[i] = bn[i]; }
  }
#pragma unroll
  for (int i = 0; i < 4; ++i)
#pragma unroll
    for (int j = 0; j < 4; ++j) c[i][j] = wmma_bf16(a[i], b[j], c[i][j]);

  const int lane = threadIdx.x;
  const int bm = (lane >> 4) << 3, cn = lane & 15;
#pragma unroll
  for (int i = 0; i < 4; ++i) {
#pragma unroll
    for (int r = 0; r < 8; ++r) {
      size_t row = (size_t)(i0 + i * 16 + bm + r) * N + j0 + cn;
      if (F32OUT) {
        float* C = (float*)Cout;
#pragma unroll
        for (int j = 0; j < 4; ++j) C[row + j * 16] = c[i][j][r];
      } else {
        uint16_t* C = (uint16_t*)Cout;
#pragma unroll
        for (int j = 0; j < 4; ++j) C[row + j * 16] = f2bf(c[i][j][r]);
      }
    }
  }
}

// ---- flash attention: one wave per (qblock16, head, batch) -----------------
__global__ __launch_bounds__(32) void attn_kernel(const uint16_t* __restrict__ qkv,
                                                  const uint16_t* __restrict__ vt,
                                                  uint16_t* __restrict__ ao) {
  __shared__ uint16_t Pl[16 * 32];  // P tile staging: C-layout -> A-layout
  const int lane = threadIdx.x;
  const int qb = blockIdx.x * 16;
  const int h  = blockIdx.y;
  const int b  = blockIdx.z;
  const int kvh = h >> 2;  // GQA 4:1

  const uint16_t* Qbase = qkv + (size_t)(b * SEQ + qb) * QKV_N + h * HEAD_DIM;
  const v16bf qa0 = load_a_tile(Qbase, QKV_N);        // dims 0..31
  const v16bf qa1 = load_a_tile(Qbase + 32, QKV_N);   // dims 32..63

  const uint16_t* Kbase = qkv + (size_t)b * SEQ * QKV_N + DIM + kvh * HEAD_DIM;
  const uint16_t* Vbase = vt + (size_t)(b * NUM_KV_HEADS + kvh) * HEAD_DIM * SEQ;

  v8f o0 = {}, o1 = {}, o2 = {}, o3 = {};
  float m_run[8], l_run[8];
#pragma unroll
  for (int r = 0; r < 8; ++r) { m_run[r] = -3.0e38f; l_run[r] = 0.0f; }

  const int bm = (lane >> 4) << 3;  // C-layout row base
  const int cn = lane & 15;         // C-layout column

  for (int j = 0; j < SEQ; j += 32) {
    // S = (Q * 1/sqrt(64)) K^T : two 16x16 score tiles (keys j..j+15, j+16..j+31)
    const uint16_t* K0 = Kbase + (size_t)j * QKV_N;
    const uint16_t* K1 = K0 + (size_t)16 * QKV_N;
    v8f s0 = {}, s1 = {};
    s0 = wmma_bf16(qa0, load_b_tile(K0, QKV_N), s0);
    s0 = wmma_bf16(qa1, load_b_tile(K0 + 32, QKV_N), s0);
    s1 = wmma_bf16(qa0, load_b_tile(K1, QKV_N), s1);
    s1 = wmma_bf16(qa1, load_b_tile(K1 + 32, QKV_N), s1);

    float corr[8];
#pragma unroll
    for (int r = 0; r < 8; ++r) {
      float a  = s0[r] * 0.125f;
      float bb = s1[r] * 0.125f;
      float rm = fmaxf(a, bb);
      rm = fmaxf(rm, __shfl_xor(rm, 1, 32));
      rm = fmaxf(rm, __shfl_xor(rm, 2, 32));
      rm = fmaxf(rm, __shfl_xor(rm, 4, 32));
      rm = fmaxf(rm, __shfl_xor(rm, 8, 32));
      float mnew = fmaxf(m_run[r], rm);
      float c  = __expf(m_run[r] - mnew);
      float p0 = __expf(a - mnew);
      float p1 = __expf(bb - mnew);
      float rs = p0 + p1;
      rs += __shfl_xor(rs, 1, 32);
      rs += __shfl_xor(rs, 2, 32);
      rs += __shfl_xor(rs, 4, 32);
      rs += __shfl_xor(rs, 8, 32);
      l_run[r] = l_run[r] * c + rs;
      m_run[r] = mnew;
      corr[r]  = c;
      Pl[(bm + r) * 32 + cn]      = f2bf(p0);
      Pl[(bm + r) * 32 + 16 + cn] = f2bf(p1);
    }
#pragma unroll
    for (int r = 0; r < 8; ++r) {
      o0[r] *= corr[r]; o1[r] *= corr[r]; o2[r] *= corr[r]; o3[r] *= corr[r];
    }
    // re-read P in A-layout (single wave: LDS pipeline is in-order per wave)
    const int pr = lane & 15, pk = (lane >> 4) << 3;
    TileU pt;
    pt.q[0] = *(const uint4*)&Pl[pr * 32 + pk];
    pt.q[1] = *(const uint4*)&Pl[pr * 32 + pk + 16];
    const v16bf pa = pt.v;
    // O += P * V   (Vt rows = head dims, contiguous over keys)
    o0 = wmma_bf16(pa, load_b_tile(Vbase + j, SEQ), o0);
    o1 = wmma_bf16(pa, load_b_tile(Vbase + (size_t)16 * SEQ + j, SEQ), o1);
    o2 = wmma_bf16(pa, load_b_tile(Vbase + (size_t)32 * SEQ + j, SEQ), o2);
    o3 = wmma_bf16(pa, load_b_tile(Vbase + (size_t)48 * SEQ + j, SEQ), o3);
  }

#pragma unroll
  for (int r = 0; r < 8; ++r) {
    float inv = 1.0f / l_run[r];
    size_t out = (size_t)(b * SEQ + qb + bm + r) * DIM + h * HEAD_DIM + cn;
    ao[out +  0] = f2bf(o0[r] * inv);
    ao[out + 16] = f2bf(o1[r] * inv);
    ao[out + 32] = f2bf(o2[r] * inv);
    ao[out + 48] = f2bf(o3[r] * inv);
  }
}

// ---- launcher --------------------------------------------------------------
extern "C" void kernel_launch(void* const* d_in, const int* in_sizes, int n_in,
                              void* d_out, int out_size, void* d_ws, size_t ws_size,
                              hipStream_t stream) {
  const float* x  = (const float*)d_in[0];
  const float* Wq = (const float*)d_in[1];
  const float* Wk = (const float*)d_in[2];
  const float* Wv = (const float*)d_in[3];
  const float* Wo = (const float*)d_in[4];
  float* out = (float*)d_out;

  uint8_t* ws = (uint8_t*)d_ws;
  uint16_t* xb    = (uint16_t*)(ws);                       // [4096 x 2048] bf16
  uint16_t* WtQKV = (uint16_t*)(ws + 16777216ull);         // [3072 x 2048] bf16 (Wq^T|Wk^T|Wv^T)
  uint16_t* WoT   = (uint16_t*)(ws + 29360128ull);         // [2048 x 2048] bf16
  uint16_t* QKV   = (uint16_t*)(ws + 37748736ull);         // [4096 x 3072] bf16
  uint16_t* Vt    = (uint16_t*)(ws + 62914560ull);         // [2 x 8 x 64 x 2048] bf16
  uint16_t* AO    = (uint16_t*)(ws + 67108864ull);         // [4096 x 2048] bf16

  // 1) convert + transpose inputs/weights to bf16
  convert_f32_bf16<<<(TOKENS * DIM / 4 + 255) / 256, 256, 0, stream>>>(x, xb, TOKENS * DIM);
  transpose_w<<<dim3(DIM / 32, DIM / 32), dim3(32, 8), 0, stream>>>(Wq, WtQKV, DIM, DIM);
  transpose_w<<<dim3(KV_DIM / 32, DIM / 32), dim3(32, 8), 0, stream>>>(
      Wk, WtQKV + (size_t)DIM * DIM, DIM, KV_DIM);
  transpose_w<<<dim3(KV_DIM / 32, DIM / 32), dim3(32, 8), 0, stream>>>(
      Wv, WtQKV + (size_t)(DIM + KV_DIM) * DIM, DIM, KV_DIM);
  transpose_w<<<dim3(DIM / 32, DIM / 32), dim3(32, 8), 0, stream>>>(Wo, WoT, DIM, DIM);

  // 2) fused QKV projection: [4096x2048] x [2048x3072]
  gemm_bf16<false><<<dim3(QKV_N / 64, TOKENS / 64), 32, 0, stream>>>(
      xb, WtQKV, QKV, TOKENS, QKV_N, DIM);

  // 3) V^T per (batch, kv-head) for contiguous PV B-operand loads
  build_vt<<<(BATCH * NUM_KV_HEADS * HEAD_DIM * SEQ) / 256, 256, 0, stream>>>(QKV, Vt);

  // 4) flash attention
  attn_kernel<<<dim3(SEQ / 16, NUM_HEADS, BATCH), 32, 0, stream>>>(QKV, Vt, AO);

  // 5) output projection -> f32
  gemm_bf16<true><<<dim3(DIM / 64, TOKENS / 64), 32, 0, stream>>>(
      AO, WoT, out, TOKENS, DIM, DIM);
}
